// S4DBlock_6253472383764
// MI455X (gfx1250) — matrix operand order repaired and checked by reading, not compiled, not run on hardware
//
#include <hip/hip_runtime.h>

#define D_MODEL 256
#define D_STATE 64
#define BATCH   4
#define SEQLEN  2048
#define CHUNK   32
#define NCHUNK  (SEQLEN / CHUNK)      // 64
#define NCOLS   (BATCH * NCHUNK)      // 256  (j = b*NCHUNK + c)
#define LN_EPS  1e-5f

typedef __attribute__((ext_vector_type(16))) _Float16 v16h;
typedef __attribute__((ext_vector_type(8)))  float    v8f;

// ---------------------------------------------------------------------------
// K1: LayerNorm over D; write xn (f32, B,L,D) and transposed f16 tiles
//     xT[d][s][j], s = l % 32, j = b*64 + l/32.   One block per (b,l).
// ---------------------------------------------------------------------------
__global__ void k_ln_transpose(const float* __restrict__ x,
                               const float* __restrict__ nw,
                               const float* __restrict__ nb,
                               float* __restrict__ xn,
                               _Float16* __restrict__ xT) {
    const int bl = blockIdx.x;              // b*SEQLEN + l
    const int b  = bl / SEQLEN;
    const int l  = bl % SEQLEN;
    const int d  = threadIdx.x;

    __shared__ float red[D_MODEL];
    float v = x[bl * D_MODEL + d];

    red[d] = v;
    __syncthreads();
    for (int off = D_MODEL / 2; off > 0; off >>= 1) {
        if (d < off) red[d] += red[d + off];
        __syncthreads();
    }
    const float mu = red[0] * (1.0f / D_MODEL);
    __syncthreads();

    const float dv = v - mu;
    red[d] = dv * dv;
    __syncthreads();
    for (int off = D_MODEL / 2; off > 0; off >>= 1) {
        if (d < off) red[d] += red[d + off];
        __syncthreads();
    }
    const float var = red[0] * (1.0f / D_MODEL);

    const float xnv = dv * rsqrtf(var + LN_EPS) * nw[d] + nb[d];
    xn[bl * D_MODEL + d] = xnv;

    const int c = l >> 5;                   // chunk
    const int s = l & 31;                   // pos in chunk
    const int j = b * NCHUNK + c;
    xT[d * (CHUNK * NCOLS) + s * NCOLS + j] = (_Float16)xnv;
}

// ---------------------------------------------------------------------------
// K2: per-channel GEMM  U_d(64x256) = W_d(64x32) * X_d(32x256)
//     W_d[n,s] = B[d,n] * dA[d,n]^(31-s).  One block (8 waves) per d.
// ---------------------------------------------------------------------------
__global__ void k_phase1(const float* __restrict__ A_log,
                         const float* __restrict__ log_dt,
                         const float* __restrict__ Bm,
                         const _Float16* __restrict__ xT,
                         float* __restrict__ U) {
    const int d   = blockIdx.x;
    const int tid = threadIdx.x;

    __shared__ _Float16 Wsh[D_STATE * CHUNK];   // 64x32
    __shared__ _Float16 Xsh[CHUNK * NCOLS];     // 32x256

    const float dtd = __expf(log_dt[d]);

    for (int idx = tid; idx < D_STATE * CHUNK; idx += 256) {
        const int n = idx >> 5, s = idx & 31;
        const float an = -__expf(A_log[n]);
        const float pw = __expf(dtd * an * (float)(31 - s));
        Wsh[idx] = (_Float16)(Bm[d * D_STATE + n] * pw);
    }
    for (int i = tid; i < CHUNK * NCOLS; i += 256)
        Xsh[i] = xT[d * (CHUNK * NCOLS) + i];
    __syncthreads();

    const int wave   = tid >> 5;
    const int lane   = tid & 31;
    const int lm     = lane & 15;
    const int laneHi = lane >> 4;

    for (int nt = 0; nt < 4; ++nt) {
        for (int jj = 0; jj < 2; ++jj) {
            const int jt = wave * 2 + jj;
            v16h a, bf;
            v8f  acc = {};
#pragma unroll
            for (int i = 0; i < 16; ++i) {
                const int ka = ((i < 8) ? i : i + 8) + laneHi * 8;
                a[i]  = Wsh[(nt * 16 + lm) * CHUNK + ka];
                const int kb = ((i < 8) ? i : i + 8) + laneHi * 8;
                bf[i] = Xsh[kb * NCOLS + jt * 16 + lm];
            }
            acc = __builtin_amdgcn_wmma_f32_16x16x32_f16(
                false, a, false, bf, (short)0, acc, false, false);
#pragma unroll
            for (int r = 0; r < 8; ++r) {
                const int n = nt * 16 + r + laneHi * 8;
                U[d * (D_STATE * NCOLS) + n * NCOLS + jt * 16 + lm] = acc[r];
            }
        }
    }
}

// ---------------------------------------------------------------------------
// K3: chunk-level scan over c (length 64), in place on U -> h_in[c].
//     65536 independent (d,n,b) sequences, decay dA^32.
// ---------------------------------------------------------------------------
__global__ void k_chunk_scan(const float* __restrict__ A_log,
                             const float* __restrict__ log_dt,
                             float* __restrict__ U) {
    const int gid = blockIdx.x * 256 + threadIdx.x;   // d*256 + n*4 + b
    const int b = gid & 3;
    const int n = (gid >> 2) & 63;
    const int d = gid >> 8;

    const float an   = -__expf(A_log[n]);
    const float dtd  = __expf(log_dt[d]);
    const float adec = __expf((float)CHUNK * dtd * an);

    float* base = U + d * (D_STATE * NCOLS) + n * NCOLS + b * NCHUNK;
    float prev = 0.0f;
    for (int c = 0; c < NCHUNK; ++c) {
        const float u = base[c];
        base[c] = prev;                  // state entering chunk c
        prev = adec * prev + u;
    }
}

// ---------------------------------------------------------------------------
// K4: Y_d(32x256) = M_d(32x32)*X_d + G_d(32x64)*Hin_d(64x256); fuse skip+res.
//     One block (8 waves) per d.
// ---------------------------------------------------------------------------
__global__ void k_phase3(const float* __restrict__ x,
                         const float* __restrict__ A_log,
                         const float* __restrict__ log_dt,
                         const float* __restrict__ Bm,
                         const float* __restrict__ Cm,
                         const float* __restrict__ Dv,
                         const float* __restrict__ xn,
                         const _Float16* __restrict__ xT,
                         const float* __restrict__ U,
                         float* __restrict__ out) {
    const int d   = blockIdx.x;
    const int tid = threadIdx.x;

    __shared__ _Float16 Gsh[CHUNK * D_STATE];   // 32x64  (4 KB)
    __shared__ _Float16 Msh[CHUNK * CHUNK];     // 32x32  (2 KB)
    __shared__ _Float16 Hsh[D_STATE * NCOLS];   // 64x256 (32 KB)
    __shared__ _Float16 Xsh[CHUNK * NCOLS];     // 32x256 (16 KB)
    __shared__ float    Kv[CHUNK];

    const float dtd = __expf(log_dt[d]);

    // G[t][n] = C[d,n] * dA^(t+1)
    for (int idx = tid; idx < CHUNK * D_STATE; idx += 256) {
        const int t = idx >> 6, n = idx & 63;
        const float an = -__expf(A_log[n]);
        Gsh[idx] = (_Float16)(Cm[d * D_STATE + n] * __expf(dtd * an * (float)(t + 1)));
    }
    // conv kernel K[tau] = sum_n C*B*dA^tau
    if (tid < CHUNK) {
        float s = 0.0f;
        for (int n = 0; n < D_STATE; ++n) {
            const float an = -__expf(A_log[n]);
            s += Cm[d * D_STATE + n] * Bm[d * D_STATE + n] *
                 __expf(dtd * an * (float)tid);
        }
        Kv[tid] = s;
    }
    for (int i = tid; i < D_STATE * NCOLS; i += 256)
        Hsh[i] = (_Float16)U[d * (D_STATE * NCOLS) + i];
    for (int i = tid; i < CHUNK * NCOLS; i += 256)
        Xsh[i] = xT[d * (CHUNK * NCOLS) + i];
    __syncthreads();

    // lower-triangular Toeplitz M[t][s] = K[t-s]
    for (int idx = tid; idx < CHUNK * CHUNK; idx += 256) {
        const int t = idx >> 5, s = idx & 31;
        Msh[idx] = (s <= t) ? (_Float16)Kv[t - s] : (_Float16)0.0f;
    }
    __syncthreads();

    const int wave   = tid >> 5;
    const int lane   = tid & 31;
    const int lm     = lane & 15;
    const int laneHi = lane >> 4;
    const float dvd  = Dv[d];

    for (int tt = 0; tt < 2; ++tt) {
        for (int jj = 0; jj < 2; ++jj) {
            const int jt = wave * 2 + jj;
            v16h a, bf;
            v8f  acc = {};

            // intra-chunk:  M (32x32) x X (32x256), K=32
#pragma unroll
            for (int i = 0; i < 16; ++i) {
                const int ka = ((i < 8) ? i : i + 8) + laneHi * 8;
                a[i]  = Msh[(tt * 16 + lm) * CHUNK + ka];
                const int kb = ((i < 8) ? i : i + 8) + laneHi * 8;
                bf[i] = Xsh[kb * NCOLS + jt * 16 + lm];
            }
            acc = __builtin_amdgcn_wmma_f32_16x16x32_f16(
                false, a, false, bf, (short)0, acc, false, false);

            // cross-chunk:  G (32x64) x Hin (64x256), K=64 => 2 WMMAs
#pragma unroll
            for (int kk = 0; kk < 2; ++kk) {
#pragma unroll
                for (int i = 0; i < 16; ++i) {
                    const int ka = ((i < 8) ? i : i + 8) + laneHi * 8;
                    a[i]  = Gsh[(tt * 16 + lm) * D_STATE + kk * 32 + ka];
                    const int kb = ((i < 8) ? i : i + 8) + laneHi * 8;
                    bf[i] = Hsh[(kk * 32 + kb) * NCOLS + jt * 16 + lm];
                }
                acc = __builtin_amdgcn_wmma_f32_16x16x32_f16(
                    false, a, false, bf, (short)0, acc, false, false);
            }

            // writeback with skip + residual
#pragma unroll
            for (int r = 0; r < 8; ++r) {
                const int t  = tt * 16 + r + laneHi * 8;
                const int j  = jt * 16 + lm;
                const int bb = j >> 6;          // batch
                const int c  = j & 63;          // chunk
                const int l  = c * CHUNK + t;
                const int gi = (bb * SEQLEN + l) * D_MODEL + d;
                out[gi] = acc[r] + xn[gi] * dvd + x[gi];
            }
        }
    }
}

// ---------------------------------------------------------------------------
extern "C" void kernel_launch(void* const* d_in, const int* in_sizes, int n_in,
                              void* d_out, int out_size, void* d_ws, size_t ws_size,
                              hipStream_t stream) {
    const float* x      = (const float*)d_in[0];
    const float* A_log  = (const float*)d_in[1];
    const float* log_dt = (const float*)d_in[2];
    const float* Bm     = (const float*)d_in[3];
    const float* Cm     = (const float*)d_in[4];
    const float* Dv     = (const float*)d_in[5];
    const float* nw     = (const float*)d_in[6];
    const float* nb     = (const float*)d_in[7];
    float*       out    = (float*)d_out;

    char* ws = (char*)d_ws;
    float*    xn = (float*)ws;                                  // 8 MB
    _Float16* xT = (_Float16*)(ws + (size_t)(8u << 20));        // 4 MB
    float*    U  = (float*)(ws + (size_t)(12u << 20));          // 16 MB

    k_ln_transpose<<<BATCH * SEQLEN, D_MODEL, 0, stream>>>(x, nw, nb, xn, xT);
    k_phase1<<<D_MODEL, 256, 0, stream>>>(A_log, log_dt, Bm, xT, U);
    k_chunk_scan<<<(D_MODEL * D_STATE * BATCH) / 256, 256, 0, stream>>>(A_log, log_dt, U);
    k_phase3<<<D_MODEL, 256, 0, stream>>>(x, A_log, log_dt, Bm, Cm, Dv, xn, xT, U, out);
}
